// MPNN_13675175870846
// MI455X (gfx1250) — matrix operand order repaired and checked by reading, not compiled, not run on hardware
//
#include <hip/hip_runtime.h>
#include <hip/hip_bf16.h>
#include <stdint.h>

#define N_NODES 20000
#define N_EDGES 100000
#define NBATCH  512
#define H       64
#define IN_CH   16
#define EDGE_DIM 8
#define H1      128      // edge-net hidden width
#define EWC     4096     // H*H columns of the big GEMM

typedef __attribute__((ext_vector_type(16))) __bf16 v16bf;
typedef __attribute__((ext_vector_type(8)))  float  v8f;

union BF16Frag { v16bf v; uint32_t d[8]; unsigned short u[16]; };

__device__ __forceinline__ unsigned short f2bf(float f) {
  union { float f; uint32_t u; } a; a.f = f;
  uint32_t u = a.u;
  uint32_t r = (u + 0x7fffu + ((u >> 16) & 1u)) >> 16;   // round-to-nearest-even
  return (unsigned short)r;
}
__device__ __forceinline__ float sigmoidf_(float x){ return 1.0f/(1.0f+expf(-x)); }

// ---------------- node embedding: out = relu(x @ lin0_w + b) -----------------
__global__ void k_lin0(const float* __restrict__ x, const float* __restrict__ w,
                       const float* __restrict__ b, float* __restrict__ out) {
  int n = blockIdx.x, t = threadIdx.x;           // block=64
  __shared__ float xr[IN_CH];
  if (t < IN_CH) xr[t] = x[n*IN_CH + t];
  __syncthreads();
  float acc = b[t];
#pragma unroll
  for (int i = 0; i < IN_CH; ++i) acc += xr[i]*w[i*H + t];
  out[(size_t)n*H + t] = fmaxf(acc, 0.f);
}

// -------- edge net layer 1: h1 = relu(edge_attr @ nn1_w + b), bf16 out -------
__global__ void k_edge_h1(const float* __restrict__ ea, const float* __restrict__ w,
                          const float* __restrict__ b, unsigned short* __restrict__ h1) {
  int id = blockIdx.x*blockDim.x + threadIdx.x;  // E*128 threads exactly
  int e = id >> 7, j = id & 127;
  const float* row = ea + (size_t)e*EDGE_DIM;
  float acc = b[j];
#pragma unroll
  for (int i = 0; i < EDGE_DIM; ++i) acc += row[i]*w[i*H1 + j];
  h1[id] = f2bf(fmaxf(acc, 0.f));
}

// ---- pre-swizzle nn2 (128x4096 fp32) into WMMA B-fragment bf16 layout -------
// B tile = 32(K) x 16(N). Contiguous k order: lanes 0-15 hold N=lane, K pairs in V0..7
// (K=0..15); lanes 16-31 hold N=lane-16, K=16..31.
__global__ void k_pack_nn2(const float* __restrict__ nn2, unsigned short* __restrict__ pb) {
  int id = blockIdx.x*blockDim.x + threadIdx.x;  // 256 ntiles * 4 ksteps * 32 lanes
  int ln = id & 31, kk = (id >> 5) & 3, nt = id >> 7;
  int n = nt*16 + (ln & 15);
  int kbase = kk*32 + (ln >> 4)*16;
  uint32_t* dst = (uint32_t*)pb + (size_t)((nt*4 + kk)*32 + ln)*8;
#pragma unroll
  for (int v = 0; v < 8; ++v) {
    int k = kbase + 2*v;
    uint32_t lo = f2bf(nn2[(size_t)k*EWC + n]);
    uint32_t hi = f2bf(nn2[(size_t)(k+1)*EWC + n]);
    dst[v] = lo | (hi << 16);
  }
}

__global__ void k_deg(const int* __restrict__ dst, float* __restrict__ deg) {
  int e = blockIdx.x*blockDim.x + threadIdx.x;
  if (e < N_EDGES) atomicAdd(&deg[dst[e]], 1.0f);
}
__global__ void k_count(const int* __restrict__ batch, int* __restrict__ cnt) {
  int n = blockIdx.x*blockDim.x + threadIdx.x;
  if (n < N_NODES) atomicAdd(&cnt[batch[n]], 1);
}
__global__ void k_scan(const int* __restrict__ cnt, int* __restrict__ start) {
  if (blockIdx.x == 0 && threadIdx.x == 0) {
    int s = 0;
    for (int b = 0; b < NBATCH; ++b) { start[b] = s; s += cnt[b]; }
  }
}

// ------------------- fused NNConv message + scatter-add ----------------------
// Per 32-edge block (4 waves, wave owns 16 output cols o0..o0+15):
//   for i in 0..63:  D(32x16) = h1_tile(32x128) x nn2_frag(128x16 @ cols
//   i*64+o) via 2x4 v_wmma_f32_16x16x32_bf16 (B fragment reused for both
//   M-tiles -> half the L2 traffic), then msg[m][o] += D[m][o]*out[src[m]][i].
// out[src] gather is staged into LDS with GLOBAL_LOAD_ASYNC_TO_LDS_B128.
__global__ void __launch_bounds__(128) k_msg(
    const unsigned short* __restrict__ h1,
    const unsigned short* __restrict__ pb,
    const float* __restrict__ nn2_b,
    const int* __restrict__ ei,           // [2,E]
    const float* __restrict__ node_h,
    float* __restrict__ agg) {
  const int bt   = blockIdx.x;            // 32-edge tile
  const int t    = threadIdx.x;           // 0..127 (4 waves)
  const int wave = t >> 5;                // o-block 0..3
  const int ln   = t & 31;
  const int half = ln >> 4;
  const int nlo  = ln & 15;
  const int o0   = wave * 16;
  const int e0   = bt * 32;

  __shared__ float a_t[32*H];             // out[src] features for 32 edges
  __shared__ int   dst_s[32];
  if (t < 32) dst_s[t] = ei[N_EDGES + e0 + t];

  // async gather: 32 rows x 64 floats = 512 x b128, 4 per thread.
  // Low 32 bits of a flat LDS pointer are the architected LDS byte offset.
#pragma unroll
  for (int c = t; c < 512; c += 128) {
    int r = c >> 4;                       // edge row 0..31
    int c4 = (c & 15) * 4;                // float column
    const float* g = node_h + (size_t)ei[e0 + r]*H + c4;
    uint32_t lds = (uint32_t)(uintptr_t)(&a_t[r*H + c4]);
    asm volatile("global_load_async_to_lds_b128 %0, %1, off"
                 :: "v"(lds), "v"(g) : "memory");
  }

  // A fragments (2 M-tiles x 4 k-steps, contiguous k = 16*half + s, same as pb)
  BF16Frag afr[2][4];
  {
    int m = ln & 15;
#pragma unroll
    for (int mt = 0; mt < 2; ++mt) {
      const unsigned short* row = h1 + (size_t)(e0 + mt*16 + m)*H1;
#pragma unroll
      for (int kk = 0; kk < 4; ++kk)
#pragma unroll
        for (int v = 0; v < 8; ++v) {
          int k = kk*32 + 16*half + 2*v;
          afr[mt][kk].d[v] = *(const uint32_t*)(row + k);
        }
    }
  }
  asm volatile("s_wait_asynccnt 0x0" ::: "memory");
  __syncthreads();

  float msg0[8], msg1[8];
#pragma unroll
  for (int r = 0; r < 8; ++r) { msg0[r] = 0.f; msg1[r] = 0.f; }

  for (int i = 0; i < H; ++i) {
    int nt = i*4 + wave;                  // column tile (i*64 + o0)/16
    float bias = nn2_b[i*H + o0 + nlo];
    v8f c0, c1;
#pragma unroll
    for (int r = 0; r < 8; ++r) { c0[r] = bias; c1[r] = bias; }
#pragma unroll
    for (int kk = 0; kk < 4; ++kk) {
      BF16Frag bfr;
      const uint4* bp = (const uint4*)pb + ((size_t)((nt*4 + kk)*32 + ln))*2;
      uint4 lo = bp[0], hi = bp[1];
      bfr.d[0]=lo.x; bfr.d[1]=lo.y; bfr.d[2]=lo.z; bfr.d[3]=lo.w;
      bfr.d[4]=hi.x; bfr.d[5]=hi.y; bfr.d[6]=hi.z; bfr.d[7]=hi.w;
      c0 = __builtin_amdgcn_wmma_f32_16x16x32_bf16(
               false, afr[0][kk].v, false, bfr.v, (short)0, c0, false, false);
      c1 = __builtin_amdgcn_wmma_f32_16x16x32_bf16(
               false, afr[1][kk].v, false, bfr.v, (short)0, c1, false, false);
    }
#pragma unroll
    for (int r = 0; r < 8; ++r) {
      msg0[r] += c0[r] * a_t[(r + 8*half)*H + i];        // scale by out[src][i]
      msg1[r] += c1[r] * a_t[(16 + r + 8*half)*H + i];
    }
  }
#pragma unroll
  for (int r = 0; r < 8; ++r) {
    int m = r + 8*half;
    atomicAdd(&agg[(size_t)dst_s[m]*H + o0 + nlo],      msg0[r]);
    atomicAdd(&agg[(size_t)dst_s[16 + m]*H + o0 + nlo], msg1[r]);
  }
}

// ----------- m = relu(agg/deg + out @ root_w + root_b) -----------------------
__global__ void k_combine(const float* __restrict__ agg, const float* __restrict__ deg,
                          const float* __restrict__ node_h, const float* __restrict__ rw,
                          const float* __restrict__ rb, float* __restrict__ mbuf) {
  int n = blockIdx.x, t = threadIdx.x;   // block=64
  __shared__ float hr[H];
  hr[t] = node_h[(size_t)n*H + t];
  __syncthreads();
  float acc = rb[t];
#pragma unroll
  for (int i = 0; i < H; ++i) acc += hr[i]*rw[i*H + t];
  float d = fmaxf(deg[n], 1.0f);
  mbuf[(size_t)n*H + t] = fmaxf(acc + agg[(size_t)n*H + t]/d, 0.f);
}

// ----------------------------- GRU cell --------------------------------------
__global__ void k_gru(const float* __restrict__ mbuf, const float* __restrict__ wih,
                      const float* __restrict__ whh, const float* __restrict__ bih,
                      const float* __restrict__ bhh, float* __restrict__ node_h) {
  int n = blockIdx.x, t = threadIdx.x;   // block=192
  __shared__ float mv[H], hv[H], gi[3*H], gh[3*H];
  if (t < H) { mv[t] = mbuf[(size_t)n*H + t]; hv[t] = node_h[(size_t)n*H + t]; }
  __syncthreads();
  float a1 = bih[t], a2 = bhh[t];
  for (int i = 0; i < H; ++i) { a1 += mv[i]*wih[i*3*H + t]; a2 += hv[i]*whh[i*3*H + t]; }
  gi[t] = a1; gh[t] = a2;
  __syncthreads();
  if (t < H) {
    float r = sigmoidf_(gi[t] + gh[t]);
    float z = sigmoidf_(gi[H + t] + gh[H + t]);
    float nn = tanhf(gi[2*H + t] + r*gh[2*H + t]);
    node_h[(size_t)n*H + t] = (1.f - z)*nn + z*hv[t];
  }
}

// ----------------------------- Set2Set LSTM ----------------------------------
__global__ void k_lstm(const float* __restrict__ qstar, const float* __restrict__ wih,
                       const float* __restrict__ whh, const float* __restrict__ bih,
                       const float* __restrict__ bhh, float* __restrict__ hh,
                       float* __restrict__ cc) {
  int b = blockIdx.x, t = threadIdx.x;   // block=256
  __shared__ float qs[2*H], hv[H], g[4*H];
  if (t < 2*H) qs[t] = qstar[(size_t)b*2*H + t];
  if (t < H)   hv[t] = hh[(size_t)b*H + t];
  __syncthreads();
  float acc = bih[t] + bhh[t];
  for (int i = 0; i < 2*H; ++i) acc += qs[i]*wih[i*4*H + t];
  for (int i = 0; i < H;   ++i) acc += hv[i]*whh[i*4*H + t];
  g[t] = acc;
  __syncthreads();
  if (t < H) {
    float ig = sigmoidf_(g[t]);
    float fg = sigmoidf_(g[H + t]);
    float gg = tanhf(g[2*H + t]);
    float og = sigmoidf_(g[3*H + t]);
    float c  = fg*cc[(size_t)b*H + t] + ig*gg;
    cc[(size_t)b*H + t] = c;
    hh[(size_t)b*H + t] = og*tanhf(c);
  }
}

// --------------------- Set2Set attention (batch is sorted) -------------------
__global__ void k_attn(const float* __restrict__ node_h, const int* __restrict__ start,
                       const int* __restrict__ cnt, const float* __restrict__ hh,
                       float* __restrict__ ebuf, float* __restrict__ qstar) {
  int b = blockIdx.x, t = threadIdx.x;   // block=64
  __shared__ float qv[H], red[H];
  qv[t] = hh[(size_t)b*H + t];
  __syncthreads();
  int s = start[b], c = cnt[b];
  // pass 1: logits + max
  float lmax = -3.402823e38f;
  for (int n = s + t; n < s + c; n += H) {
    const float* row = node_h + (size_t)n*H;
    float e = 0.f;
#pragma unroll
    for (int i = 0; i < H; ++i) e += row[i]*qv[i];
    ebuf[n] = e;
    lmax = fmaxf(lmax, e);
  }
  red[t] = lmax; __syncthreads();
  for (int sft = 32; sft > 0; sft >>= 1) { if (t < sft) red[t] = fmaxf(red[t], red[t+sft]); __syncthreads(); }
  float emax = red[0]; __syncthreads();
  // pass 2: exp + sum
  float ls = 0.f;
  for (int n = s + t; n < s + c; n += H) {
    float a = expf(ebuf[n] - emax);
    ebuf[n] = a; ls += a;
  }
  red[t] = ls; __syncthreads();
  for (int sft = 32; sft > 0; sft >>= 1) { if (t < sft) red[t] += red[t+sft]; __syncthreads(); }
  float ssum = red[0]; __syncthreads();
  // pass 3: weighted sum, component o = t (coalesced across lanes)
  float rv = 0.f;
  for (int n = s; n < s + c; ++n) rv += ebuf[n]*node_h[(size_t)n*H + t];
  qstar[(size_t)b*2*H + t]     = qv[t];
  qstar[(size_t)b*2*H + H + t] = (c > 0) ? rv/ssum : 0.f;
}

// -------------------------------- head ---------------------------------------
__global__ void k_head(const float* __restrict__ qstar, const float* __restrict__ w1,
                       const float* __restrict__ b1, const float* __restrict__ w2,
                       const float* __restrict__ b2, float* __restrict__ out) {
  int b = blockIdx.x, t = threadIdx.x;   // block=64
  __shared__ float qs[2*H], red[H];
  qs[t] = qstar[(size_t)b*2*H + t];
  qs[H + t] = qstar[(size_t)b*2*H + H + t];
  __syncthreads();
  float acc = b1[t];
  for (int i = 0; i < 2*H; ++i) acc += qs[i]*w1[i*H + t];
  red[t] = fmaxf(acc, 0.f)*w2[t];
  __syncthreads();
  for (int sft = 32; sft > 0; sft >>= 1) { if (t < sft) red[t] += red[t+sft]; __syncthreads(); }
  if (t == 0) out[b] = red[0] + b2[0];
}

extern "C" void kernel_launch(void* const* d_in, const int* in_sizes, int n_in,
                              void* d_out, int out_size, void* d_ws, size_t ws_size,
                              hipStream_t stream) {
  const float* x       = (const float*)d_in[0];
  const int*   ei      = (const int*)  d_in[1];
  const float* eattr   = (const float*)d_in[2];
  const int*   batch   = (const int*)  d_in[3];
  const float* lin0_w  = (const float*)d_in[4];
  const float* lin0_b  = (const float*)d_in[5];
  const float* nn1_w   = (const float*)d_in[6];
  const float* nn1_b   = (const float*)d_in[7];
  const float* nn2_w   = (const float*)d_in[8];
  const float* nn2_b   = (const float*)d_in[9];
  const float* root_w  = (const float*)d_in[10];
  const float* root_b  = (const float*)d_in[11];
  const float* gru_wih = (const float*)d_in[12];
  const float* gru_whh = (const float*)d_in[13];
  const float* gru_bih = (const float*)d_in[14];
  const float* gru_bhh = (const float*)d_in[15];
  const float* lstm_wih= (const float*)d_in[16];
  const float* lstm_whh= (const float*)d_in[17];
  const float* lstm_bih= (const float*)d_in[18];
  const float* lstm_bhh= (const float*)d_in[19];
  const float* lin1_w  = (const float*)d_in[20];
  const float* lin1_b  = (const float*)d_in[21];
  const float* lin2_w  = (const float*)d_in[22];
  const float* lin2_b  = (const float*)d_in[23];
  float* out = (float*)d_out;

  char* base = (char*)d_ws;
  size_t off = 0;
  auto take = [&](size_t bytes) -> char* {
    char* r = base + off;
    off = (off + bytes + 255) & ~(size_t)255;
    return r;
  };
  unsigned short* h1    = (unsigned short*)take((size_t)N_EDGES*H1*2);       // 25.6 MB bf16
  unsigned short* pb    = (unsigned short*)take((size_t)256*4*32*16*2);      // 1 MB packed nn2
  float* node_h = (float*)take((size_t)N_NODES*H*4);
  float* agg    = (float*)take((size_t)N_NODES*H*4);
  float* mbuf   = (float*)take((size_t)N_NODES*H*4);
  float* deg    = (float*)take((size_t)N_NODES*4);
  float* ebuf   = (float*)take((size_t)N_NODES*4);
  float* qstar  = (float*)take((size_t)NBATCH*2*H*4);
  float* hh     = (float*)take((size_t)NBATCH*H*4);
  float* cc     = (float*)take((size_t)NBATCH*H*4);
  int*   cnt    = (int*)  take((size_t)NBATCH*4);
  int*   start  = (int*)  take((size_t)NBATCH*4);

  // ---- prep ----
  hipMemsetAsync(deg, 0, (size_t)N_NODES*4, stream);
  hipMemsetAsync(cnt, 0, (size_t)NBATCH*4, stream);
  k_lin0   <<<N_NODES, 64, 0, stream>>>(x, lin0_w, lin0_b, node_h);
  k_edge_h1<<<(N_EDGES*H1)/256, 256, 0, stream>>>(eattr, nn1_w, nn1_b, h1);
  k_pack_nn2<<<(256*4*32)/256, 256, 0, stream>>>(nn2_w, pb);
  k_deg    <<<(N_EDGES+255)/256, 256, 0, stream>>>(ei + N_EDGES, deg);
  k_count  <<<(N_NODES+255)/256, 256, 0, stream>>>(batch, cnt);
  k_scan   <<<1, 1, 0, stream>>>(cnt, start);

  // ---- 3 message-passing + GRU steps (fused WMMA NNConv, ew never stored) ----
  for (int step = 0; step < 3; ++step) {
    hipMemsetAsync(agg, 0, (size_t)N_NODES*H*4, stream);
    k_msg    <<<N_EDGES/32, 128, 0, stream>>>(h1, pb, nn2_b, ei, node_h, agg);
    k_combine<<<N_NODES, 64, 0, stream>>>(agg, deg, node_h, root_w, root_b, mbuf);
    k_gru    <<<N_NODES, 192, 0, stream>>>(mbuf, gru_wih, gru_whh, gru_bih, gru_bhh, node_h);
  }

  // ---- Set2Set (3 steps) ----
  hipMemsetAsync(qstar, 0, (size_t)NBATCH*2*H*4, stream);
  hipMemsetAsync(hh, 0, (size_t)NBATCH*H*4, stream);
  hipMemsetAsync(cc, 0, (size_t)NBATCH*H*4, stream);
  for (int it = 0; it < 3; ++it) {
    k_lstm<<<NBATCH, 256, 0, stream>>>(qstar, lstm_wih, lstm_whh, lstm_bih, lstm_bhh, hh, cc);
    k_attn<<<NBATCH, 64, 0, stream>>>(node_h, start, cnt, hh, ebuf, qstar);
  }

  // ---- head ----
  k_head<<<NBATCH, 64, 0, stream>>>(qstar, lin1_w, lin1_b, lin2_w, lin2_b, out);
}
